// Model_91182155694223
// MI455X (gfx1250) — matrix operand order repaired and checked by reading, not compiled, not run on hardware
//
#include <hip/hip_runtime.h>

#define HD 768
#define BN_EPS 1e-5f
#define LN_EPS 1e-12f
#define LP 40   // LDS row pitch in ushorts (80B: conflict-free b128 frag loads)

typedef __attribute__((ext_vector_type(16))) __bf16          v16bf;
typedef __attribute__((ext_vector_type(8)))  float           v8f;
typedef __attribute__((ext_vector_type(8)))  unsigned short  v8us;

__device__ __forceinline__ unsigned short f2bf(float f) {
  return __builtin_bit_cast(unsigned short, (__bf16)f);   // v_cvt_bf16_f32 (RNE)
}

union frag_cast { v8us h[2]; v16bf v; };

__device__ __forceinline__ v16bf ld_frag(const unsigned short* p) {
  frag_cast u;
  u.h[0] = *(const v8us*)(p);        // ds_load_b128
  u.h[1] = *(const v8us*)(p + 8);    // ds_load_b128
  return u.v;
}

// ---------------------------------------------------------------------------
// Tiled GEMM: Y[M,N] = A[M,K] @ W[K,N] + bias   (fp32 in/out, bf16 WMMA core)
// Block tile 64x128, 8 waves (wave32); each wave: 32x32 (4 16x16 WMMA tiles).
// Double-buffered LDS (one barrier per K-step). All global staging loads use
// immediate offsets from a single base (2x b128 for A, 4x b128 for B per thread), so
// there are no dependent address chains / xcnt stalls; the transpose of B is
// done by 16 ds_store_u16 with immediate offsets.
// A stored [row][k], k contiguous, so an A-fragment is 32 contiguous bytes;
// B stored transposed [col][k] so a B-fragment is 32 contiguous bytes.
// Requires M%64==0, N%128==0, K%32==0 (true for every GEMM in this model).
// ---------------------------------------------------------------------------
__global__ __launch_bounds__(256)
void gemm_bf16(const float* __restrict__ A, const float* __restrict__ W,
               const float* __restrict__ bias, float* __restrict__ Y,
               int M, int N, int K, long long lda, long long ldy)
{
  __shared__ unsigned short As[2][64][LP];    // [buf][row][k]
  __shared__ unsigned short Bt[2][128][LP];   // [buf][col][k]

  const int tid  = threadIdx.x;
  const int lane = tid & 31;
  const int wave = tid >> 5;
  const int wm   = (wave >> 2) * 32;        // wave M offset in block tile
  const int wn   = (wave & 3) * 32;         // wave N offset in block tile
  const long long bM = (long long)blockIdx.y * 64;
  const long long bN = (long long)blockIdx.x * 128;

  // A staging: 64 rows x 32 k, 8 consecutive floats/thread,
  // dest k contiguous (same order as Bt)
  const int ar  = tid >> 2;
  const int ac  = (tid & 3) * 8;
  const int apc = ac;
  // B staging: 32 k-rows x 128 cols; thread owns 16 consecutive cols of one
  // k-row (4x b128 global loads), scattered to Bt[col][k] via immediate-offset
  // ds_store_u16 (transpose in the store pattern, no address chains)
  const int bkr = tid >> 3;          // k row 0..31
  const int bcg = (tid & 7) * 16;    // col group base

  const int r = lane & 15, hh = lane >> 4;

  v8f c00 = {}, c01 = {}, c10 = {}, c11 = {};
  float areg[8];
  float breg[16];

  // ---- prologue: stage tile 0 into buffer 0 ----
  {
    const float* ap = A + (bM + ar) * lda + ac;
    const float* bp = W + (long long)bkr * N + (bN + bcg);
#pragma unroll
    for (int j = 0; j < 8; ++j) areg[j] = ap[j];
#pragma unroll
    for (int j = 0; j < 16; ++j) breg[j] = bp[j];
  }
#pragma unroll
  for (int j = 0; j < 8; ++j) As[0][ar][apc + j] = f2bf(areg[j]);
#pragma unroll
  for (int j = 0; j < 16; ++j) Bt[0][bcg + j][bkr] = f2bf(breg[j]);
  __syncthreads();

  int buf = 0;
  for (int k0 = 32; k0 < K; k0 += 32) {
    // issue global loads of tile k0 (overlap with WMMAs below)
    const float* ap = A + (bM + ar) * lda + (k0 + ac);
    const float* bp = W + (long long)(k0 + bkr) * N + (bN + bcg);
    if (k0 + 32 < K) {                       // keep GL2 warm one tile ahead
      __builtin_prefetch(ap + 32, 0, 3);
      __builtin_prefetch(bp + 32LL * N, 0, 3);
    }
#pragma unroll
    for (int j = 0; j < 8; ++j) areg[j] = ap[j];
#pragma unroll
    for (int j = 0; j < 16; ++j) breg[j] = bp[j];

    // compute on current buffer
    {
      const v16bf a0 = ld_frag(&As[buf][wm + r][hh * 16]);
      const v16bf a1 = ld_frag(&As[buf][wm + 16 + r][hh * 16]);
      const v16bf b0 = ld_frag(&Bt[buf][wn + r][hh * 16]);
      const v16bf b1 = ld_frag(&Bt[buf][wn + 16 + r][hh * 16]);
      c00 = __builtin_amdgcn_wmma_f32_16x16x32_bf16(false, a0, false, b0, (short)0, c00, false, false);
      c01 = __builtin_amdgcn_wmma_f32_16x16x32_bf16(false, a0, false, b1, (short)0, c01, false, false);
      c10 = __builtin_amdgcn_wmma_f32_16x16x32_bf16(false, a1, false, b0, (short)0, c10, false, false);
      c11 = __builtin_amdgcn_wmma_f32_16x16x32_bf16(false, a1, false, b1, (short)0, c11, false, false);
    }

    // stage tile k0 into the other buffer; single barrier per K-step
    const int nb = buf ^ 1;
#pragma unroll
    for (int j = 0; j < 8; ++j) As[nb][ar][apc + j] = f2bf(areg[j]);
#pragma unroll
    for (int j = 0; j < 16; ++j) Bt[nb][bcg + j][bkr] = f2bf(breg[j]);
    __syncthreads();
    buf = nb;
  }

  // ---- epilogue: last tile ----
  {
    const v16bf a0 = ld_frag(&As[buf][wm + r][hh * 16]);
    const v16bf a1 = ld_frag(&As[buf][wm + 16 + r][hh * 16]);
    const v16bf b0 = ld_frag(&Bt[buf][wn + r][hh * 16]);
    const v16bf b1 = ld_frag(&Bt[buf][wn + 16 + r][hh * 16]);
    c00 = __builtin_amdgcn_wmma_f32_16x16x32_bf16(false, a0, false, b0, (short)0, c00, false, false);
    c01 = __builtin_amdgcn_wmma_f32_16x16x32_bf16(false, a0, false, b1, (short)0, c01, false, false);
    c10 = __builtin_amdgcn_wmma_f32_16x16x32_bf16(false, a1, false, b0, (short)0, c10, false, false);
    c11 = __builtin_amdgcn_wmma_f32_16x16x32_bf16(false, a1, false, b1, (short)0, c11, false, false);
  }

  const long long n0 = bN + wn + r;
  const long long n1 = n0 + 16;
  const float bv0 = bias[n0];
  const float bv1 = bias[n1];
  const long long m0 = bM + wm + hh * 8;    // C/D layout: VGPR rr -> M = rr + 8*(lane>=16)
#pragma unroll
  for (int rr = 0; rr < 8; ++rr) {
    Y[(m0 + rr)      * ldy + n0] = c00[rr] + bv0;
    Y[(m0 + rr)      * ldy + n1] = c01[rr] + bv1;
    Y[(m0 + rr + 16) * ldy + n0] = c10[rr] + bv0;
    Y[(m0 + rr + 16) * ldy + n1] = c11[rr] + bv1;
  }
}

// -------------------- BatchNorm (training-mode batch stats) -----------------
__global__ __launch_bounds__(256)
void colstats(const float* __restrict__ Y, float* __restrict__ mean,
              float* __restrict__ var, int M, int N, long long pitch)
{
  __shared__ float sS[8][32], sQ[8][32];
  const int c  = threadIdx.x & 31;
  const int ry = threadIdx.x >> 5;
  const long long col = (long long)blockIdx.x * 32 + c;
  float s = 0.f, q = 0.f;
  for (int m = ry; m < M; m += 8) {
    float v = Y[(long long)m * pitch + col];
    s += v; q += v * v;
  }
  sS[ry][c] = s; sQ[ry][c] = q;
  __syncthreads();
  if (ry == 0) {
#pragma unroll
    for (int j = 1; j < 8; ++j) { s += sS[j][c]; q += sQ[j][c]; }
    const float mu = s / (float)M;
    mean[col] = mu;
    var[col]  = q / (float)M - mu * mu;   // biased, like torch BN
  }
}

__global__ void bn_act(float* __restrict__ Y, const float* __restrict__ g,
                       const float* __restrict__ be, const float* __restrict__ mean,
                       const float* __restrict__ var, long long M, long long N,
                       long long pitch, int leaky)
{
  long long idx = (long long)blockIdx.x * blockDim.x + threadIdx.x;
  if (idx >= M * N) return;
  const long long m = idx / N, n = idx % N;
  float* p = Y + m * pitch + n;
  float y = g[n] * (*p - mean[n]) * rsqrtf(var[n] + BN_EPS) + be[n];
  if (leaky && y < 0.f) y *= 0.2f;
  *p = y;
}

// -------------------- elementwise / fusion helpers --------------------------
__global__ void masked_mean_k(const float* __restrict__ x0, const float* __restrict__ x1,
                              const float* __restrict__ x2, const int* __restrict__ s,
                              float* __restrict__ c, long long B)
{
  long long idx = (long long)blockIdx.x * blockDim.x + threadIdx.x;
  if (idx >= B * HD) return;
  const long long b = idx / HD;
  const float s0 = (float)s[b * 3 + 0];
  const float s1 = (float)s[b * 3 + 1];
  const float s2 = (float)s[b * 3 + 2];
  c[idx] = (s0 * x0[idx] + s1 * x1[idx] + s2 * x2[idx]) / (s0 + s1 + s2);
}

__global__ void fill_where_k(const float* __restrict__ x, const float* __restrict__ xr,
                             const int* __restrict__ s, int view,
                             float* __restrict__ outp, long long B, long long D)
{
  long long idx = (long long)blockIdx.x * blockDim.x + threadIdx.x;
  if (idx >= B * D) return;
  const long long b = idx / D;
  outp[idx] = (s[b * 3 + view] > 0) ? x[idx] : xr[idx];
}

__global__ __launch_bounds__(256)
void l2norm_rows(const float* __restrict__ src, float* __restrict__ dst,
                 int D, long long spitch, long long dpitch)
{
  __shared__ float red[256];
  const long long row = blockIdx.x;
  const float* sp = src + row * spitch;
  float q = 0.f;
  for (int i = threadIdx.x; i < D; i += 256) { float v = sp[i]; q += v * v; }
  red[threadIdx.x] = q; __syncthreads();
  for (int t = 128; t > 0; t >>= 1) {
    if (threadIdx.x < t) red[threadIdx.x] += red[threadIdx.x + t];
    __syncthreads();
  }
  const float inv = 1.f / fmaxf(sqrtf(red[0]), 1e-12f);
  float* dp = dst + row * dpitch;
  for (int i = threadIdx.x; i < D; i += 256) dp[i] = sp[i] * inv;
}

__global__ void fill_cls_k(float* __restrict__ seq, const float* __restrict__ cls, long long B)
{
  long long idx = (long long)blockIdx.x * blockDim.x + threadIdx.x;
  if (idx >= B * HD) return;
  const long long b = idx / HD;
  const int h = (int)(idx % HD);
  seq[(b * 4) * HD + h] = cls[h];
}

// -------------------- LayerNorm variants ------------------------------------
__global__ __launch_bounds__(256)
void embed_ln(float* __restrict__ hbuf, const float* __restrict__ pos,
              const float* __restrict__ typ, const float* __restrict__ g,
              const float* __restrict__ be)
{
  __shared__ float r1[256], r2[256];
  const long long row = blockIdx.x;
  const int t = (int)(row & 3);
  float* hp = hbuf + row * HD;
  float v[3]; float s = 0.f, q = 0.f;
#pragma unroll
  for (int j = 0; j < 3; ++j) {
    const int i = threadIdx.x + j * 256;
    v[j] = hp[i] + pos[t * HD + i] + typ[t * HD + i];
    s += v[j]; q += v[j] * v[j];
  }
  r1[threadIdx.x] = s; r2[threadIdx.x] = q; __syncthreads();
  for (int st = 128; st > 0; st >>= 1) {
    if (threadIdx.x < st) { r1[threadIdx.x] += r1[threadIdx.x + st]; r2[threadIdx.x] += r2[threadIdx.x + st]; }
    __syncthreads();
  }
  const float mu  = r1[0] / (float)HD;
  const float inv = rsqrtf(r2[0] / (float)HD - mu * mu + LN_EPS);
#pragma unroll
  for (int j = 0; j < 3; ++j) {
    const int i = threadIdx.x + j * 256;
    hp[i] = g[i] * (v[j] - mu) * inv + be[i];
  }
}

__global__ __launch_bounds__(256)
void residual_ln(const float* __restrict__ x, const float* __restrict__ z,
                 const float* __restrict__ g, const float* __restrict__ be,
                 float* __restrict__ outp)
{
  __shared__ float r1[256], r2[256];
  const long long row = blockIdx.x;
  const float* xp = x + row * HD;
  const float* zp = z + row * HD;
  float v[3]; float s = 0.f, q = 0.f;
#pragma unroll
  for (int j = 0; j < 3; ++j) {
    const int i = threadIdx.x + j * 256;
    v[j] = xp[i] + zp[i];
    s += v[j]; q += v[j] * v[j];
  }
  r1[threadIdx.x] = s; r2[threadIdx.x] = q; __syncthreads();
  for (int st = 128; st > 0; st >>= 1) {
    if (threadIdx.x < st) { r1[threadIdx.x] += r1[threadIdx.x + st]; r2[threadIdx.x] += r2[threadIdx.x + st]; }
    __syncthreads();
  }
  const float mu  = r1[0] / (float)HD;
  const float inv = rsqrtf(r2[0] / (float)HD - mu * mu + LN_EPS);
  float* op = outp + row * HD;
#pragma unroll
  for (int j = 0; j < 3; ++j) {
    const int i = threadIdx.x + j * 256;
    op[i] = g[i] * (v[j] - mu) * inv + be[i];
  }
}

// -------------------- tiny attention (L=4, dh=64, 12 heads) -----------------
__global__ void attention4(const float* __restrict__ Q, const float* __restrict__ Kx,
                           const float* __restrict__ V, const int* __restrict__ s,
                           float* __restrict__ ctx, long long B)
{
  long long idx = (long long)blockIdx.x * blockDim.x + threadIdx.x;
  if (idx >= B * 48) return;
  const long long b = idx / 48;
  const int rem = (int)(idx % 48);
  const int hd = rem >> 2;
  const int i  = rem & 3;
  const float* qp = Q + (b * 4 + i) * HD + hd * 64;
  float sc[4];
#pragma unroll
  for (int j = 0; j < 4; ++j) {
    const float* kp = Kx + (b * 4 + j) * HD + hd * 64;
    float d = 0.f;
    for (int t = 0; t < 64; ++t) d += qp[t] * kp[t];
    const float maskadd = (j == 0 || s[b * 3 + (j - 1)] > 0) ? 0.f : -10000.f;
    sc[j] = d * 0.125f + maskadd;     // 1/sqrt(64)
  }
  const float mx = fmaxf(fmaxf(sc[0], sc[1]), fmaxf(sc[2], sc[3]));
  float ps = 0.f;
#pragma unroll
  for (int j = 0; j < 4; ++j) { sc[j] = __expf(sc[j] - mx); ps += sc[j]; }
  const float inv = 1.f / ps;
  float* op = ctx + (b * 4 + i) * HD + hd * 64;
  for (int t = 0; t < 64; ++t) {
    float a = 0.f;
#pragma unroll
    for (int j = 0; j < 4; ++j) a += sc[j] * V[(b * 4 + j) * HD + hd * 64 + t];
    op[t] = a * inv;
  }
}

// -------------------- misc elementwise --------------------------------------
__global__ void gelu_k(float* __restrict__ p, long long n) {
  long long idx = (long long)blockIdx.x * blockDim.x + threadIdx.x;
  if (idx >= n) return;
  const float x = p[idx];
  p[idx] = 0.5f * x * (1.f + erff(x * 0.70710678118654752f));
}
__global__ void tanh_k(float* __restrict__ p, long long n) {
  long long idx = (long long)blockIdx.x * blockDim.x + threadIdx.x;
  if (idx >= n) return;
  p[idx] = tanhf(p[idx]);
}
__global__ void logits_k(const float* __restrict__ P, const float* __restrict__ W,
                         const float* __restrict__ b, float* __restrict__ outp, long long B)
{
  long long idx = (long long)blockIdx.x * blockDim.x + threadIdx.x;
  if (idx >= B * 10) return;
  const long long bb = idx / 10;
  const int c = (int)(idx % 10);
  float acc = b[c];
  const float* pr = P + bb * HD;
  for (int k = 0; k < HD; ++k) acc += pr[k] * W[k * 10 + c];
  outp[idx] = acc;
}

// ---------------------------------------------------------------------------
extern "C" void kernel_launch(void* const* d_in, const int* in_sizes, int n_in,
                              void* d_out, int out_size, void* d_ws, size_t ws_size,
                              hipStream_t stream)
{
  (void)in_sizes; (void)out_size; (void)ws_size;
  const long long B = 8192;
  const int VD[3] = {1024, 2048, 512};

  const float* X[3] = {(const float*)d_in[0], (const float*)d_in[1], (const float*)d_in[2]};
  const int* s = (const int*)d_in[3];
  const float* P[93] = {};
  for (int i = 0; i < 93 && i < n_in; ++i) P[i] = (const float*)d_in[i];

  // param leaf indices (dict insertion order flattening of setup_inputs)
  const int I_CLS = 4;
  auto projIdx = [](int v) { return 5 + v * 8; };        // W0,b0,g0,be0,W1,b1,g1,be1
  auto genIdx  = [](int v) { return 29 + v * 8; };
  const int I_POS = 53, I_TYPE = 54, I_LNG = 55, I_LNB = 56;
  auto layIdx  = [](int l) { return 57 + l * 16; };      // Wq,bq,Wk,bk,Wv,bv,Wo,bo,ln1g,ln1b,Wi,bi,Wf,bf,ln2g,ln2b
  const int I_POOLW = 89, I_POOLB = 90, I_CLFW = 91, I_CLFB = 92;

  // output layout: logits | xr0..2 | xm0..2 | features
  float* out = (float*)d_out;
  float* out_logits = out;
  float* out_xr[3]; float* out_xm[3];
  size_t oo = (size_t)B * 10;
  for (int v = 0; v < 3; ++v) { out_xr[v] = out + oo; oo += (size_t)B * VD[v]; }
  for (int v = 0; v < 3; ++v) { out_xm[v] = out + oo; oo += (size_t)B * VD[v]; }
  float* out_feat = out + oo;

  // workspace
  float* w = (float*)d_ws;
  size_t off = 0;
  auto take = [&](size_t nf) { float* p = w + off; off += (nf + 63) & ~(size_t)63; return p; };
  float* xp[3]; for (int v = 0; v < 3; ++v) xp[v] = take((size_t)B * HD);
  float* hid  = take((size_t)B * 2048);
  float* xf   = take((size_t)B * 2048);
  float* cbuf = take((size_t)B * HD);
  float* H0 = take((size_t)4 * B * HD);   // seq / h
  float* H1 = take((size_t)4 * B * HD);   // a
  float* Qb = take((size_t)4 * B * HD);   // Q, then attn-out projection
  float* Kb = take((size_t)4 * B * HD);   // K, then FF hidden (8192x3072)
  float* Vb = take((size_t)4 * B * HD);   // V, then FF output
  float* Cx = take((size_t)4 * B * HD);   // attention context
  float* d_mean = take(4096);
  float* d_var  = take(4096);

  auto gemm = [&](const float* A, long long lda, const float* Wt, const float* bias,
                  float* Y, long long ldy, int M, int N, int K) {
    dim3 g((unsigned)(N / 128), (unsigned)(M / 64));
    gemm_bf16<<<g, 256, 0, stream>>>(A, Wt, bias, Y, M, N, K, lda, ldy);
  };
  auto bnorm = [&](float* Y, long long pitch, int M, int N,
                   const float* g, const float* be, int leaky) {
    colstats<<<dim3((unsigned)(N / 32)), 256, 0, stream>>>(Y, d_mean, d_var, M, N, pitch);
    long long tot = (long long)M * N;
    bn_act<<<dim3((unsigned)((tot + 255) / 256)), 256, 0, stream>>>(
        Y, g, be, d_mean, d_var, (long long)M, (long long)N, pitch, leaky);
  };
  auto ew = [&](long long n) { return dim3((unsigned)((n + 255) / 256)); };

  // ---- proj MLPs on raw views -> xp[v] ----
  for (int v = 0; v < 3; ++v) {
    const int pb = projIdx(v);
    gemm(X[v], VD[v], P[pb + 0], P[pb + 1], hid, 1024, (int)B, 1024, VD[v]);
    bnorm(hid, 1024, (int)B, 1024, P[pb + 2], P[pb + 3], 1);
    gemm(hid, 1024, P[pb + 4], P[pb + 5], xp[v], HD, (int)B, HD, 1024);
    bnorm(xp[v], HD, (int)B, HD, P[pb + 6], P[pb + 7], 0);
  }

  // ---- masked mean over present views -> cbuf ----
  masked_mean_k<<<ew(B * HD), 256, 0, stream>>>(xp[0], xp[1], xp[2], s, cbuf, B);

  // ---- gen MLPs -> xr (directly into d_out), then fill/xm/xp2 per view ----
  for (int v = 0; v < 3; ++v) {
    const int gb = genIdx(v);
    gemm(cbuf, HD, P[gb + 0], P[gb + 1], hid, 1024, (int)B, 1024, HD);
    bnorm(hid, 1024, (int)B, 1024, P[gb + 2], P[gb + 3], 1);
    gemm(hid, 1024, P[gb + 4], P[gb + 5], out_xr[v], VD[v], (int)B, VD[v], 1024);
    bnorm(out_xr[v], VD[v], (int)B, VD[v], P[gb + 6], P[gb + 7], 0);

    // xfill = observed ? x : generated
    fill_where_k<<<ew(B * VD[v]), 256, 0, stream>>>(X[v], out_xr[v], s, v, xf, B, VD[v]);
    // xm = l2n(xfill)
    l2norm_rows<<<dim3((unsigned)B), 256, 0, stream>>>(xf, out_xm[v], VD[v], VD[v], VD[v]);

    // xp2 = proj MLP on xfill, written straight into seq token (1+v)
    const int pb = projIdx(v);
    gemm(xf, VD[v], P[pb + 0], P[pb + 1], hid, 1024, (int)B, 1024, VD[v]);
    bnorm(hid, 1024, (int)B, 1024, P[pb + 2], P[pb + 3], 1);
    float* seqTok = H0 + (size_t)(1 + v) * HD;            // row pitch 4*HD
    gemm(hid, 1024, P[pb + 4], P[pb + 5], seqTok, 4 * HD, (int)B, HD, 1024);
    bnorm(seqTok, 4 * HD, (int)B, HD, P[pb + 6], P[pb + 7], 0);

    // features[:, v, :] = l2n(seq token) before BERT mutates it
    l2norm_rows<<<dim3((unsigned)B), 256, 0, stream>>>(seqTok, out_feat + (size_t)v * HD,
                                                       HD, 4 * HD, 3 * HD);
  }

  // ---- BERT ----
  fill_cls_k<<<ew(B * HD), 256, 0, stream>>>(H0, P[I_CLS], B);
  embed_ln<<<dim3((unsigned)(4 * B)), 256, 0, stream>>>(H0, P[I_POS], P[I_TYPE],
                                                        P[I_LNG], P[I_LNB]);
  const int Mt = (int)(4 * B);   // 32768 token rows
  for (int l = 0; l < 2; ++l) {
    const int lb = layIdx(l);
    gemm(H0, HD, P[lb + 0], P[lb + 1], Qb, HD, Mt, HD, HD);   // Q
    gemm(H0, HD, P[lb + 2], P[lb + 3], Kb, HD, Mt, HD, HD);   // K
    gemm(H0, HD, P[lb + 4], P[lb + 5], Vb, HD, Mt, HD, HD);   // V
    attention4<<<ew(B * 48), 256, 0, stream>>>(Qb, Kb, Vb, s, Cx, B);
    gemm(Cx, HD, P[lb + 6], P[lb + 7], Qb, HD, Mt, HD, HD);   // attn output proj -> Qb
    residual_ln<<<dim3((unsigned)Mt), 256, 0, stream>>>(H0, Qb, P[lb + 8], P[lb + 9], H1);

    for (int c = 0; c < 4; ++c) {     // FF in 8192-row chunks, hidden in Kb
      const float* abase = H1 + (size_t)c * B * HD;
      gemm(abase, HD, P[lb + 10], P[lb + 11], Kb, 3072, (int)B, 3072, HD);
      gelu_k<<<ew(B * 3072), 256, 0, stream>>>(Kb, B * 3072);
      gemm(Kb, 3072, P[lb + 12], P[lb + 13], Vb + (size_t)c * B * HD, HD, (int)B, HD, 3072);
    }
    residual_ln<<<dim3((unsigned)Mt), 256, 0, stream>>>(H1, Vb, P[lb + 14], P[lb + 15], H0);
  }

  // ---- pooler + classifier ----
  gemm(H0, 4 * HD, P[I_POOLW], P[I_POOLB], hid, HD, (int)B, HD, HD);  // h[:,0] via lda
  tanh_k<<<ew(B * HD), 256, 0, stream>>>(hid, B * HD);
  logits_k<<<ew(B * 10), 256, 0, stream>>>(hid, P[I_CLFW], P[I_CLFB], out_logits, B);
}
